// MHABlock_15822659518598
// MI455X (gfx1250) — matrix operand order repaired and checked by reading, not compiled, not run on hardware
//
#include <hip/hip_runtime.h>
#include <hip/hip_bf16.h>

typedef __bf16 bf16_t;
typedef __attribute__((ext_vector_type(16))) __bf16 v16bf;
typedef __attribute__((ext_vector_type(8)))  __bf16 v8bf;
typedef __attribute__((ext_vector_type(4)))  __bf16 v4bf;
typedef __attribute__((ext_vector_type(8)))  float  v8f;
typedef __attribute__((ext_vector_type(4)))  float  v4f;
typedef __attribute__((ext_vector_type(4)))  int    v4i;

#define DEV __device__ __forceinline__

#if __has_builtin(__builtin_amdgcn_global_load_async_to_lds_b128)
#define USE_ASYNC_LDS 1
#else
#define USE_ASYNC_LDS 0
#endif

// builtin signature (from hipcc diagnostic): (v4i AS1*, v4i AS3*, imm, imm)
typedef __attribute__((address_space(1))) v4i* gv4_t;
typedef __attribute__((address_space(3))) v4i* lv4_t;

union V16U { v16bf v; v8bf h[2]; };

DEV v8f vzero8() {
  v8f z;
#pragma unroll
  for (int i = 0; i < 8; ++i) z[i] = 0.0f;
  return z;
}

template <int N>
DEV void wait_async() {
#if __has_builtin(__builtin_amdgcn_s_wait_asynccnt)
  __builtin_amdgcn_s_wait_asynccnt(N);
#else
  asm volatile("s_wait_asynccnt %0" ::"i"(N) : "memory");
#endif
}

// ---------------------------------------------------------------------------
// Core bf16 WMMA GEMM tile: C[BM,BN] (+)= A[m0:m0+BM, 0:K] * Bt[n0:n0+BN, 0:K]^T
// A: row-major [*, lda] bf16 (reduction dim contiguous)
// Bt: N-major  [*, ldb] bf16 (reduction dim contiguous)
// Accumulators returned in WMMA C/D layout (v8f per 16x16 tile).
// ---------------------------------------------------------------------------
template <int BM, int BN, int WAVES_M, int WAVES_N, int TM, int TN>
DEV void wmma_gemm(const bf16_t* __restrict__ A, int lda,
                   const bf16_t* __restrict__ Bt, int ldb,
                   int K, int m0, int n0, v8f (&acc)[TM][TN]) {
  constexpr int KB = 32;                 // K per step (one wmma_f32_16x16x32_bf16)
  constexpr int NT = WAVES_M * WAVES_N * 32;
  constexpr int PITCH = KB + 8;          // 40 bf16 = 80 B row pitch (16B aligned)
  constexpr int CA = (BM * (KB / 8)) / NT;
  constexpr int CB = (BN * (KB / 8)) / NT;
  static_assert(BM == WAVES_M * TM * 16 && BN == WAVES_N * TN * 16, "tile cfg");
  static_assert((BM * 4) % NT == 0 && (BN * 4) % NT == 0, "loader cfg");

  const int tid = threadIdx.x;
  const int lane = tid & 31;
  const int wid = tid >> 5;
  const int wr = (wid / WAVES_N) * (TM * 16);
  const int wc = (wid % WAVES_N) * (TN * 16);
  const int l15 = lane & 15;
  const int half = lane >> 4;
  const int nk = K / KB;

#if USE_ASYNC_LDS
  // -------- async global->LDS DMA path, double-buffered LDS ---------------
  __shared__ bf16_t sA[2][BM][PITCH];
  __shared__ bf16_t sB[2][BN][PITCH];

  auto issue = [&](int koff, int buf) {
#pragma unroll
    for (int c = 0; c < CA; ++c) {
      int ch = c * NT + tid;
      int row = ch >> 2, kc = (ch & 3) << 3;
      __builtin_amdgcn_global_load_async_to_lds_b128(
          (gv4_t)(void*)(const_cast<bf16_t*>(A) + (size_t)(m0 + row) * lda + koff + kc),
          (lv4_t)(void*)&sA[buf][row][kc], 0, 0);
    }
#pragma unroll
    for (int c = 0; c < CB; ++c) {
      int ch = c * NT + tid;
      int row = ch >> 2, kc = (ch & 3) << 3;
      __builtin_amdgcn_global_load_async_to_lds_b128(
          (gv4_t)(void*)(const_cast<bf16_t*>(Bt) + (size_t)(n0 + row) * ldb + koff + kc),
          (lv4_t)(void*)&sB[buf][row][kc], 0, 0);
    }
  };

  issue(0, 0);
  for (int kb = 0; kb < nk; ++kb) {
    const int buf = kb & 1;
    if (kb + 1 < nk) {
      issue((kb + 1) * KB, buf ^ 1);   // overlap next tile DMA with compute
      wait_async<CA + CB>();           // tile kb complete (in-order returns)
    } else {
      wait_async<0>();
    }
    __syncthreads();
    if (kb + 2 < nk) {                 // GL2 prefetch two tiles ahead
      __builtin_prefetch(A + (size_t)(m0 + (tid >> 2)) * lda + (size_t)(kb + 2) * KB, 0, 0);
      __builtin_prefetch(Bt + (size_t)(n0 + (tid >> 2)) * ldb + (size_t)(kb + 2) * KB, 0, 0);
    }

    V16U af[TM];
#pragma unroll
    for (int tm = 0; tm < TM; ++tm) {
      int r = wr + tm * 16 + l15;
      af[tm].h[0] = *reinterpret_cast<const v8bf*>(&sA[buf][r][half * 8]);
      af[tm].h[1] = *reinterpret_cast<const v8bf*>(&sA[buf][r][16 + half * 8]);
    }
    V16U bfm[TN];
#pragma unroll
    for (int tn = 0; tn < TN; ++tn) {
      int r = wc + tn * 16 + l15;
      bfm[tn].h[0] = *reinterpret_cast<const v8bf*>(&sB[buf][r][half * 8]);
      bfm[tn].h[1] = *reinterpret_cast<const v8bf*>(&sB[buf][r][16 + half * 8]);
    }
#pragma unroll
    for (int tm = 0; tm < TM; ++tm)
#pragma unroll
      for (int tn = 0; tn < TN; ++tn)
        acc[tm][tn] = __builtin_amdgcn_wmma_f32_16x16x32_bf16(
            false, af[tm].v, false, bfm[tn].v, (short)0, acc[tm][tn], false, false);
    __syncthreads();                   // protect buf against kb+2 issue (WAR)
  }
#else
  // -------- fallback: register-staged pipeline ----------------------------
  __shared__ bf16_t sA[BM][PITCH];
  __shared__ bf16_t sB[BN][PITCH];
  v8bf ra[CA], rb[CB];

  auto loadG = [&](int koff) {
#pragma unroll
    for (int c = 0; c < CA; ++c) {
      int ch = c * NT + tid;
      int row = ch >> 2, kc = (ch & 3) << 3;
      ra[c] = *reinterpret_cast<const v8bf*>(A + (size_t)(m0 + row) * lda + koff + kc);
    }
#pragma unroll
    for (int c = 0; c < CB; ++c) {
      int ch = c * NT + tid;
      int row = ch >> 2, kc = (ch & 3) << 3;
      rb[c] = *reinterpret_cast<const v8bf*>(Bt + (size_t)(n0 + row) * ldb + koff + kc);
    }
  };
  auto storeS = [&]() {
#pragma unroll
    for (int c = 0; c < CA; ++c) {
      int ch = c * NT + tid;
      int row = ch >> 2, kc = (ch & 3) << 3;
      *reinterpret_cast<v8bf*>(&sA[row][kc]) = ra[c];
    }
#pragma unroll
    for (int c = 0; c < CB; ++c) {
      int ch = c * NT + tid;
      int row = ch >> 2, kc = (ch & 3) << 3;
      *reinterpret_cast<v8bf*>(&sB[row][kc]) = rb[c];
    }
  };

  loadG(0);
  for (int kb = 0; kb < nk; ++kb) {
    storeS();
    __syncthreads();
    if (kb + 1 < nk) loadG((kb + 1) * KB);
    if (kb + 2 < nk) {
      __builtin_prefetch(A + (size_t)(m0 + (tid >> 2)) * lda + (size_t)(kb + 2) * KB, 0, 0);
      __builtin_prefetch(Bt + (size_t)(n0 + (tid >> 2)) * ldb + (size_t)(kb + 2) * KB, 0, 0);
    }
    V16U af[TM];
#pragma unroll
    for (int tm = 0; tm < TM; ++tm) {
      int r = wr + tm * 16 + l15;
      af[tm].h[0] = *reinterpret_cast<const v8bf*>(&sA[r][half * 8]);
      af[tm].h[1] = *reinterpret_cast<const v8bf*>(&sA[r][16 + half * 8]);
    }
    V16U bfm[TN];
#pragma unroll
    for (int tn = 0; tn < TN; ++tn) {
      int r = wc + tn * 16 + l15;
      bfm[tn].h[0] = *reinterpret_cast<const v8bf*>(&sB[r][half * 8]);
      bfm[tn].h[1] = *reinterpret_cast<const v8bf*>(&sB[r][16 + half * 8]);
    }
#pragma unroll
    for (int tm = 0; tm < TM; ++tm)
#pragma unroll
      for (int tn = 0; tn < TN; ++tn)
        acc[tm][tn] = __builtin_amdgcn_wmma_f32_16x16x32_bf16(
            false, af[tm].v, false, bfm[tn].v, (short)0, acc[tm][tn], false, false);
    __syncthreads();
  }
#endif
}

// ---------------------------------------------------------------------------
// Prep kernels
// ---------------------------------------------------------------------------
__global__ void k_cvt_bf16(const float* __restrict__ in, bf16_t* __restrict__ out, int n4) {
  int i = blockIdx.x * blockDim.x + threadIdx.x;
  if (i < n4) {
    v4f v = *reinterpret_cast<const v4f*>(in + (size_t)i * 4);
    v4bf o;
#pragma unroll
    for (int j = 0; j < 4; ++j) o[j] = (bf16_t)v[j];
    *reinterpret_cast<v4bf*>(out + (size_t)i * 4) = o;
  }
}

// out[z*outZ + c*R + r] = (bf16) in[z*inZ + r*ld + c]
__global__ __launch_bounds__(256) void k_tr_cvt(const float* __restrict__ in,
                                                bf16_t* __restrict__ out,
                                                int R, int C, int ld,
                                                size_t inZ, size_t outZ) {
  __shared__ float tile[32][33];
  const size_t ib = (size_t)blockIdx.z * inZ;
  const size_t ob = (size_t)blockIdx.z * outZ;
  const int r0 = blockIdx.y * 32, c0 = blockIdx.x * 32;
  const int tx = threadIdx.x & 31, ty = threadIdx.x >> 5;
#pragma unroll
  for (int it = 0; it < 4; ++it) {
    int rr = ty + it * 8;
    tile[rr][tx] = in[ib + (size_t)(r0 + rr) * ld + c0 + tx];
  }
  __syncthreads();
#pragma unroll
  for (int it = 0; it < 4; ++it) {
    int cc = ty + it * 8;
    out[ob + (size_t)(c0 + cc) * R + r0 + tx] = (bf16_t)tile[tx][cc];
  }
}

// ---------------------------------------------------------------------------
// GEMM1: [4096,1024] x [1024,2048] -> Q (row-major) and K^T ([b,h,c,j])
// ---------------------------------------------------------------------------
__global__ __launch_bounds__(256) void k_gemm_qk(const bf16_t* __restrict__ xb,
                                                 const bf16_t* __restrict__ wqkt,
                                                 const float* __restrict__ bqkv,
                                                 bf16_t* __restrict__ qb,
                                                 bf16_t* __restrict__ kt) {
  constexpr int BM = 128, BN = 128, WM = 4, WN = 2, TM = 2, TN = 4;
  const int m0 = blockIdx.y * BM, n0 = blockIdx.x * BN;
  v8f acc[TM][TN];
#pragma unroll
  for (int a = 0; a < TM; ++a)
#pragma unroll
    for (int b = 0; b < TN; ++b) acc[a][b] = vzero8();

  wmma_gemm<BM, BN, WM, WN, TM, TN>(xb, 1024, wqkt, 1024, 1024, m0, n0, acc);

  const int tid = threadIdx.x, lane = tid & 31, wid = tid >> 5;
  const int wr = (wid / WN) * (TM * 16), wc = (wid % WN) * (TN * 16);
  const int l15 = lane & 15, half = lane >> 4;
#pragma unroll
  for (int tm = 0; tm < TM; ++tm)
#pragma unroll
    for (int tn = 0; tn < TN; ++tn) {
      int col = n0 + wc + tn * 16 + l15;
      float bias = bqkv[col];
#pragma unroll
      for (int r = 0; r < 8; ++r) {
        int row = m0 + wr + tm * 16 + half * 8 + r;
        float v = acc[tm][tn][r] + bias;
        if (col < 1024) {
          qb[(size_t)row * 1024 + col] = (bf16_t)v;
        } else {
          int n2 = col - 1024, h = n2 >> 6, c = n2 & 63;
          int b = row >> 11, i = row & 2047;
          kt[((size_t)((b * 16 + h) * 64 + c)) * 2048 + i] = (bf16_t)v;
        }
      }
    }
}

// ---------------------------------------------------------------------------
// GEMM2 (per b,h): T = (1/8) * K^T[64,2048] @ Wfc_h[2048,1024], stored transposed
// Tt[b][d][h*64+c]
// ---------------------------------------------------------------------------
__global__ __launch_bounds__(256) void k_gemm_t(const bf16_t* __restrict__ kt,
                                                const bf16_t* __restrict__ wfct,
                                                bf16_t* __restrict__ Tt) {
  constexpr int BM = 64, BN = 128, WM = 2, WN = 4, TM = 2, TN = 2;
  const int bh = blockIdx.y;
  const int b = bh >> 4, h = bh & 15;
  const int n0 = blockIdx.x * BN;
  const bf16_t* A = kt + (size_t)bh * 64 * 2048;
  const bf16_t* Bt = wfct + (size_t)h * 1024 * 2048;
  v8f acc[TM][TN];
#pragma unroll
  for (int a = 0; a < TM; ++a)
#pragma unroll
    for (int c = 0; c < TN; ++c) acc[a][c] = vzero8();

  wmma_gemm<BM, BN, WM, WN, TM, TN>(A, 2048, Bt, 2048, 2048, 0, n0, acc);

  const int tid = threadIdx.x, lane = tid & 31, wid = tid >> 5;
  const int wr = (wid / WN) * (TM * 16), wc = (wid % WN) * (TN * 16);
  const int l15 = lane & 15, half = lane >> 4;
#pragma unroll
  for (int tm = 0; tm < TM; ++tm)
#pragma unroll
    for (int tn = 0; tn < TN; ++tn) {
      int d = n0 + wc + tn * 16 + l15;
#pragma unroll
      for (int r = 0; r < 8; ++r) {
        int c = wr + tm * 16 + half * 8 + r;  // 0..63
        Tt[((size_t)b * 1024 + d) * 1024 + h * 64 + c] = (bf16_t)(acc[tm][tn][r] * 0.125f);
      }
    }
}

// ---------------------------------------------------------------------------
// GEMM3 (per b): out_tmp = Q[2048,1024] @ T_flat[1024,1024]   (fp32 result)
// ---------------------------------------------------------------------------
__global__ __launch_bounds__(256) void k_gemm_out(const bf16_t* __restrict__ qb,
                                                  const bf16_t* __restrict__ Tt,
                                                  float* __restrict__ tmp) {
  constexpr int BM = 128, BN = 128, WM = 4, WN = 2, TM = 2, TN = 4;
  const int b = blockIdx.z;
  const int m0 = blockIdx.y * BM, n0 = blockIdx.x * BN;
  const bf16_t* A = qb + (size_t)b * 2048 * 1024;
  const bf16_t* Bt = Tt + (size_t)b * 1024 * 1024;
  v8f acc[TM][TN];
#pragma unroll
  for (int a = 0; a < TM; ++a)
#pragma unroll
    for (int c = 0; c < TN; ++c) acc[a][c] = vzero8();

  wmma_gemm<BM, BN, WM, WN, TM, TN>(A, 1024, Bt, 1024, 1024, m0, n0, acc);

  const int tid = threadIdx.x, lane = tid & 31, wid = tid >> 5;
  const int wr = (wid / WN) * (TM * 16), wc = (wid % WN) * (TN * 16);
  const int l15 = lane & 15, half = lane >> 4;
#pragma unroll
  for (int tm = 0; tm < TM; ++tm)
#pragma unroll
    for (int tn = 0; tn < TN; ++tn) {
      int col = n0 + wc + tn * 16 + l15;
#pragma unroll
      for (int r = 0; r < 8; ++r) {
        int row = m0 + wr + tm * 16 + half * 8 + r;
        tmp[((size_t)b * 2048 + row) * 1024 + col] = acc[tm][tn][r];
      }
    }
}

// ---------------------------------------------------------------------------
// Residual + bias + LayerNorm over D=1024 (one block per row, wave32 shuffles)
// ---------------------------------------------------------------------------
__global__ __launch_bounds__(256) void k_ln(const float* __restrict__ x,
                                            const float* __restrict__ tmp,
                                            const float* __restrict__ bfc,
                                            const float* __restrict__ gamma,
                                            const float* __restrict__ beta,
                                            float* __restrict__ out) {
  const int row = blockIdx.x;
  const size_t base = (size_t)row * 1024;
  const int t = threadIdx.x;
  float y[4];
  float s = 0.f, s2 = 0.f;
#pragma unroll
  for (int j = 0; j < 4; ++j) {
    int d = t + j * 256;
    float v = x[base + d] + tmp[base + d] + bfc[d];
    y[j] = v;
    s += v;
    s2 += v * v;
  }
#pragma unroll
  for (int off = 16; off; off >>= 1) {
    s += __shfl_xor(s, off, 32);
    s2 += __shfl_xor(s2, off, 32);
  }
  __shared__ float rs[8], rs2[8];
  const int wid = t >> 5, lane = t & 31;
  if (lane == 0) { rs[wid] = s; rs2[wid] = s2; }
  __syncthreads();
  if (wid == 0) {
    float a = lane < 8 ? rs[lane] : 0.f;
    float b = lane < 8 ? rs2[lane] : 0.f;
#pragma unroll
    for (int off = 4; off; off >>= 1) {
      a += __shfl_xor(a, off, 32);
      b += __shfl_xor(b, off, 32);
    }
    if (lane == 0) { rs[0] = a; rs2[0] = b; }
  }
  __syncthreads();
  const float mean = rs[0] * (1.0f / 1024.0f);
  const float var = rs2[0] * (1.0f / 1024.0f) - mean * mean;
  const float inv = rsqrtf(var + 1e-5f);
#pragma unroll
  for (int j = 0; j < 4; ++j) {
    int d = t + j * 256;
    out[base + d] = gamma[d] * (y[j] - mean) * inv + beta[d];
  }
}

// ---------------------------------------------------------------------------
extern "C" void kernel_launch(void* const* d_in, const int* in_sizes, int n_in,
                              void* d_out, int out_size, void* d_ws, size_t ws_size,
                              hipStream_t stream) {
  const float* x     = (const float*)d_in[0];
  const float* w_qkv = (const float*)d_in[1];
  const float* b_qkv = (const float*)d_in[2];
  const float* w_fc  = (const float*)d_in[3];
  const float* b_fc  = (const float*)d_in[4];
  const float* gamma = (const float*)d_in[5];
  const float* beta  = (const float*)d_in[6];
  float* out = (float*)d_out;

  char* ws = (char*)d_ws;
  size_t off = 0;
  auto walloc = [&](size_t bytes) -> void* {
    void* p = ws + off;
    off += (bytes + 255) & ~(size_t)255;
    return p;
  };
  bf16_t* xb   = (bf16_t*)walloc((size_t)4096 * 1024 * 2);        // x in bf16
  bf16_t* wqkt = (bf16_t*)walloc((size_t)2048 * 1024 * 2);        // w_qkv[:, :2048]^T
  bf16_t* wfct = (bf16_t*)walloc((size_t)16 * 1024 * 2048 * 2);   // w_fc as [h][d][j]
  bf16_t* qb   = (bf16_t*)walloc((size_t)2 * 2048 * 1024 * 2);    // Q bf16 [b,i,h*64+c]
  bf16_t* kt   = (bf16_t*)walloc((size_t)2 * 16 * 64 * 2048 * 2); // K^T [b,h,c,j]
  bf16_t* Tt   = (bf16_t*)walloc((size_t)2 * 1024 * 1024 * 2);    // T^T [b,d,h*64+c]
  float*  tmp  = (float*)walloc((size_t)2 * 2048 * 1024 * 4);     // fc output fp32

  // prep
  k_cvt_bf16<<<4096, 256, 0, stream>>>(x, xb, (4096 * 1024) / 4);
  k_tr_cvt<<<dim3(64, 32, 1), 256, 0, stream>>>(w_qkv, wqkt, 1024, 2048, 3072,
                                                (size_t)0, (size_t)0);
  k_tr_cvt<<<dim3(32, 64, 16), 256, 0, stream>>>(w_fc, wfct, 2048, 1024, 16 * 1024,
                                                 (size_t)1024, (size_t)1024 * 2048);
  // Q/K projection
  k_gemm_qk<<<dim3(16, 32), 256, 0, stream>>>(xb, wqkt, b_qkv, qb, kt);
  // T = (1/8) K^T @ Wfc_h  (32 independent skinny GEMMs)
  k_gemm_t<<<dim3(8, 32), 256, 0, stream>>>(kt, wfct, Tt);
  // out = Q @ T_flat
  k_gemm_out<<<dim3(8, 16, 2), 256, 0, stream>>>(qb, Tt, tmp);
  // residual + bias + LayerNorm
  k_ln<<<4096, 256, 0, stream>>>(x, tmp, b_fc, gamma, beta, out);
}
